// NemotronHMTP_12481174962826
// MI455X (gfx1250) — compile-verified, matrix operand order checked by reading, not run on hardware
//
#include <hip/hip_runtime.h>
#include <hip/hip_bf16.h>

// ---------------- problem constants (match reference) ----------------
constexpr int Bc = 2, Sc = 512, Hc = 2048;
constexpr int Tc = Bc * Sc;              // 1024 tokens
constexpr int Ec = 64, Gc = 8, TOPKG = 4, Kc = 6;
constexpr int Ic = 512;                  // moe_intermediate_size
constexpr int SHI = 2048;                // shared intermediate
constexpr float RSF = 2.5f;

// ---------------- bf16 helpers (storage-only, no __bf16 arithmetic) ---
typedef __attribute__((ext_vector_type(16))) __bf16 v16bf;
typedef __attribute__((ext_vector_type(8)))  float  v8f;

// pack truncated-bf16(lo), truncated-bf16(hi) into one dword with one v_perm_b32
__device__ __forceinline__ unsigned pk2(float lo, float hi) {
    return __builtin_amdgcn_perm(__builtin_bit_cast(unsigned, hi),
                                 __builtin_bit_cast(unsigned, lo),
                                 0x07060302u);
}

__device__ __forceinline__ __bf16 f2bf(float f) {   // RNE, used only on cold paths
    unsigned u = __builtin_bit_cast(unsigned, f);
    unsigned r = u + 0x7FFFu + ((u >> 16) & 1u);
    unsigned short s = (unsigned short)(r >> 16);
    return __builtin_bit_cast(__bf16, s);
}

// A/B fragment for v_wmma_f32_16x16x32_bf16, operand row-major along K.
// Lane holds row (lane&15); hi = lane>>4 selects K sub-pattern:
//   element i in 0..7  -> k = hi*8 + i ; element i in 8..15 -> k = 16 + hi*8 + (i-8)
// i.e. two contiguous 8-element runs at p + hi*8 and p + 16 + hi*8.
struct D8 { unsigned d[8]; };
__device__ __forceinline__ v16bf frag_from_f32(const float* p, int hi) {
    const float4* q1 = (const float4*)(p + hi * 8);
    const float4* q2 = (const float4*)(p + 16 + hi * 8);
    float4 a = q1[0], b = q1[1], c = q2[0], d = q2[1];
    D8 u;
    u.d[0] = pk2(a.x, a.y); u.d[1] = pk2(a.z, a.w);
    u.d[2] = pk2(b.x, b.y); u.d[3] = pk2(b.z, b.w);
    u.d[4] = pk2(c.x, c.y); u.d[5] = pk2(c.z, c.w);
    u.d[6] = pk2(d.x, d.y); u.d[7] = pk2(d.z, d.w);
    return __builtin_bit_cast(v16bf, u);
}

struct U32B { uint4 a, b; };
__device__ __forceinline__ v16bf frag_from_bf16(const __bf16* p, int hi) {
    U32B u;
    u.a = *(const uint4*)(p + hi * 8);
    u.b = *(const uint4*)(p + 16 + hi * 8);
    return __builtin_bit_cast(v16bf, u);
}

__device__ __forceinline__ v8f wmma_bf16(v16bf a, v16bf b, v8f c) {
    return __builtin_amdgcn_wmma_f32_16x16x32_bf16(false, a, false, b,
                                                   (short)0, c, false, false);
}

// ---------------- CDNA5 async copy: global -> LDS, 16B per lane ----------
// ISA 15.18.3 op 98: GLOBAL_LOAD_ASYNC_TO_LDS_B128, tracked by ASYNCcnt.
// VDST = per-lane LDS byte address, VADDR = per-lane 64-bit global address.
__device__ __forceinline__ void async_load_b128(unsigned lds_addr, const void* gaddr) {
    asm volatile("global_load_async_to_lds_b128 %0, %1, off"
                 :: "v"(lds_addr), "v"(gaddr)
                 : "memory");
}
__device__ __forceinline__ void wait_async0() {
    asm volatile("s_wait_asynccnt 0" ::: "memory");
}

// ---------------- one-shot fp32 -> bf16 conversion ----------------
// n must be a multiple of 8; each thread converts 8 contiguous elements.
__global__ void cvt_f32_bf16_kernel(const float* __restrict__ src,
                                    __bf16* __restrict__ dst, int n) {
    int i = (blockIdx.x * blockDim.x + threadIdx.x) * 8;
    if (i >= n) return;
    const float4* s = (const float4*)(src + i);
    float4 a = s[0], b = s[1];
    uint4 o;
    o.x = pk2(a.x, a.y); o.y = pk2(a.z, a.w);
    o.z = pk2(b.x, b.y); o.w = pk2(b.z, b.w);
    *(uint4*)(dst + i) = o;
}

// ---------------- gate + routing (fp32, exact) ----------------
// grid = T blocks, block = 64 threads (one per expert)
__global__ void gate_kernel(const float* __restrict__ x,
                            const float* __restrict__ gate_w,
                            const float* __restrict__ gate_bias,
                            int* __restrict__ ecnt,
                            int* __restrict__ etok,
                            float* __restrict__ ewt) {
    const int t = blockIdx.x;
    const int e = threadIdx.x;
    const float4* xr = (const float4*)(x + (size_t)t * Hc);
    const float4* wr = (const float4*)(gate_w + (size_t)e * Hc);
    float acc = 0.f;
    for (int k = 0; k < Hc / 4; ++k) {
        float4 xv = xr[k], wv = wr[k];
        acc += xv.x * wv.x + xv.y * wv.y + xv.z * wv.z + xv.w * wv.w;
    }
    float s = 1.f / (1.f + expf(-acc));
    __shared__ float sc[Ec], sb[Ec];
    sc[e] = s;
    sb[e] = s + gate_bias[e];
    __syncthreads();
    if (e == 0) {
        float gsc[Gc];
        for (int g = 0; g < Gc; ++g) {
            float m1 = -1e30f, m2 = -1e30f;
            for (int j = 0; j < Ec / Gc; ++j) {
                float v = sb[g * (Ec / Gc) + j];
                if (v > m1) { m2 = m1; m1 = v; } else if (v > m2) { m2 = v; }
            }
            gsc[g] = m1 + m2;
        }
        unsigned gmask = 0;
        for (int r = 0; r < TOPKG; ++r) {
            int best = 0; float bv = -1e30f;
            for (int g = 0; g < Gc; ++g)
                if (!((gmask >> g) & 1u) && gsc[g] > bv) { bv = gsc[g]; best = g; }
            gmask |= 1u << best;
        }
        unsigned long long chosen = 0ull;
        int idxs[Kc]; float sum = 0.f;
        for (int k = 0; k < Kc; ++k) {
            int best = 0; float bv = -1e30f;
            for (int ee = 0; ee < Ec; ++ee) {
                if (!((gmask >> (ee / (Ec / Gc))) & 1u)) continue;
                if ((chosen >> ee) & 1ull) continue;
                if (sb[ee] > bv) { bv = sb[ee]; best = ee; }
            }
            chosen |= 1ull << best;
            idxs[k] = best;
            sum += sc[best];
        }
        float inv = RSF / (sum + 1e-20f);
        for (int k = 0; k < Kc; ++k) {
            int ee = idxs[k];
            int pos = atomicAdd(&ecnt[ee], 1);
            etok[ee * Tc + pos] = t;
            ewt[ee * Tc + pos]  = sc[ee] * inv;
        }
    }
}

// ---------------- shared-expert up: act = relu2(xbf @ wupbf^T) -> bf16 ----
// grid = (SHI/64, T/16), block = 128 (4 waves, one 16x16 tile each)
__global__ void shared_up_kernel(const __bf16* __restrict__ xbf,
                                 const __bf16* __restrict__ wupbf,
                                 __bf16* __restrict__ act) {
    const int wave = threadIdx.x >> 5, lane = threadIdx.x & 31;
    const int r = lane & 15, hi = lane >> 4;
    const int n0 = blockIdx.x * 64 + wave * 16;
    const int m0 = blockIdx.y * 16;
    const __bf16* arow = xbf   + (size_t)(m0 + r) * Hc;
    const __bf16* brow = wupbf + (size_t)(n0 + r) * Hc;
    v8f acc = {};
#pragma unroll 4
    for (int k0 = 0; k0 < Hc; k0 += 32) {
        v16bf a = frag_from_bf16(arow + k0, hi);
        v16bf b = frag_from_bf16(brow + k0, hi);
        acc = wmma_bf16(a, b, acc);
    }
    const int n = n0 + r;
#pragma unroll
    for (int rr = 0; rr < 8; ++rr) {
        int m = m0 + 8 * hi + rr;
        float v = acc[rr];
        v = v > 0.f ? v * v : 0.f;            // relu2
        act[(size_t)m * SHI + n] = f2bf(v);
    }
}

// ---------------- shared-expert down: out = act @ wdnbf^T (writes out) -----
// grid = (H/64, T/16), block = 128
__global__ void shared_down_kernel(const __bf16* __restrict__ act,
                                   const __bf16* __restrict__ wdnbf,
                                   float* __restrict__ out) {
    const int wave = threadIdx.x >> 5, lane = threadIdx.x & 31;
    const int r = lane & 15, hi = lane >> 4;
    const int n0 = blockIdx.x * 64 + wave * 16;
    const int m0 = blockIdx.y * 16;
    const __bf16* arow = act   + (size_t)(m0 + r) * SHI;
    const __bf16* brow = wdnbf + (size_t)(n0 + r) * SHI;
    v8f acc = {};
#pragma unroll 4
    for (int k0 = 0; k0 < SHI; k0 += 32) {
        v16bf a = frag_from_bf16(arow + k0, hi);
        v16bf b = frag_from_bf16(brow + k0, hi);
        acc = wmma_bf16(a, b, acc);
    }
    const int n = n0 + r;
#pragma unroll
    for (int rr = 0; rr < 8; ++rr) {
        int m = m0 + 8 * hi + rr;
        out[(size_t)m * Hc + n] = acc[rr];    // plain store; routed adds later
    }
}

// ---------------- routed experts -----------------------------------------
// grid = (T/16 token tiles, E experts), block = 256 (8 waves)
// Gathered 16-row x tile async-staged into LDS once (ASYNCcnt path), A-frags
// then come from ds_load_b128; fp32 expert weights streamed from HBM on the
// VMEM pipe, converted in-register with v_perm, prefetched ahead.
__global__ void routed_kernel(const __bf16* __restrict__ xbf,
                              const float* __restrict__ w1,
                              const float* __restrict__ w2,
                              const int*  __restrict__ ecnt,
                              const int*  __restrict__ etok,
                              const float* __restrict__ ewt,
                              float* __restrict__ out) {
    const int e  = blockIdx.y;
    const int t0 = blockIdx.x * 16;
    const int cnt = ecnt[e];
    if (t0 >= cnt) return;

    // row strides chosen so stride ≡ 4 dwords (mod 64 banks): conflict-free
    __shared__ __bf16 xtile[16][Hc + 8];    // 64.2 KB gathered A tile
    __shared__ __bf16 hbuf[16][Ic + 8];     // 16.3 KB intermediate
    __shared__ int   toks[16];
    __shared__ float wts[16];
    if (threadIdx.x < 16) {
        int j = t0 + (int)threadIdx.x;
        int jj = j < cnt ? j : cnt - 1;
        toks[threadIdx.x] = etok[e * Tc + jj];
        wts[threadIdx.x]  = (j < cnt) ? ewt[e * Tc + j] : 0.f;
    }
    __syncthreads();

    // ---- async-stage gathered x rows into LDS: 16 rows x 256 16B chunks ----
    for (int c = threadIdx.x; c < 16 * 256; c += 256) {
        const int row = c >> 8, chunk = c & 255;
        const __bf16* g = xbf + (size_t)toks[row] * Hc + chunk * 8;
        unsigned lds = (unsigned)(uintptr_t)&xtile[row][chunk * 8];
        async_load_b128(lds, g);
    }
    wait_async0();
    __syncthreads();

    const int wave = threadIdx.x >> 5, lane = threadIdx.x & 31;
    const int r = lane & 15, hi = lane >> 4;
    const __bf16* arow = &xtile[r][0];
    const float*  w1e  = w1 + (size_t)e * Ic * Hc;

    // ---- up: 32 N-tiles of 16 over I, 8 waves -> 4 tiles each ----
    for (int tile = wave; tile < Ic / 16; tile += 8) {
        const int n0 = tile * 16;
        const float* brow = w1e + (size_t)(n0 + r) * Hc;
        v8f acc = {};
#pragma unroll 4
        for (int k0 = 0; k0 < Hc; k0 += 32) {
            __builtin_prefetch(brow + k0 + 512, 0, 1);   // stream weights ahead
            v16bf a = frag_from_bf16(arow + k0, hi);
            v16bf b = frag_from_f32(brow + k0, hi);
            acc = wmma_bf16(a, b, acc);
        }
        const int n = n0 + r;
#pragma unroll
        for (int rr = 0; rr < 8; ++rr) {
            float v = acc[rr];
            v = v > 0.f ? v * v : 0.f;
            hbuf[8 * hi + rr][n] = f2bf(v);
        }
    }
    __syncthreads();

    // ---- down: 128 N-tiles of 16 over H, 8 waves -> 16 tiles each ----
    const float* w2e = w2 + (size_t)e * Hc * Ic;
    const __bf16* ar = &hbuf[r][0];
    for (int tile = wave; tile < Hc / 16; tile += 8) {
        const int n0 = tile * 16;
        const float* brow = w2e + (size_t)(n0 + r) * Ic;
        v8f acc = {};
#pragma unroll 4
        for (int k0 = 0; k0 < Ic; k0 += 32) {
            __builtin_prefetch(brow + k0 + 256, 0, 1);
            v16bf a = frag_from_bf16(ar + k0, hi);
            v16bf b = frag_from_f32(brow + k0, hi);
            acc = wmma_bf16(a, b, acc);
        }
        const int n = n0 + r;
#pragma unroll
        for (int rr = 0; rr < 8; ++rr) {
            int m = 8 * hi + rr;
            atomicAdd(&out[(size_t)toks[m] * Hc + n], wts[m] * acc[rr]);
        }
    }
}

// ---------------- launch ---------------------------------------------------
extern "C" void kernel_launch(void* const* d_in, const int* in_sizes, int n_in,
                              void* d_out, int out_size, void* d_ws, size_t ws_size,
                              hipStream_t stream) {
    const float* x      = (const float*)d_in[0];   // [B,S,H]
    const float* gate_w = (const float*)d_in[1];   // [E,H]
    const float* gbias  = (const float*)d_in[2];   // [E]
    const float* w1     = (const float*)d_in[3];   // [E,I,H]
    const float* w2     = (const float*)d_in[4];   // [E,H,I]
    const float* shup   = (const float*)d_in[5];   // [SHI,H]
    const float* shdn   = (const float*)d_in[6];   // [H,SHI]
    float* out = (float*)d_out;

    // workspace layout (all regions 16B aligned)
    int*    ecnt  = (int*)d_ws;                       // E ints
    int*    etok  = ecnt + Ec;                        // E*T ints
    float*  ewt   = (float*)(etok + Ec * Tc);         // E*T floats
    __bf16* act   = (__bf16*)(ewt + Ec * Tc);         // T*SHI bf16
    __bf16* xbf   = act + (size_t)Tc * SHI;           // T*H bf16
    __bf16* wupbf = xbf + (size_t)Tc * Hc;            // SHI*H bf16
    __bf16* wdnbf = wupbf + (size_t)SHI * Hc;         // H*SHI bf16

    hipMemsetAsync(ecnt, 0, Ec * sizeof(int), stream);

    // one-shot bf16 conversions of reused operands
    {
        int n1 = Tc * Hc, n2 = SHI * Hc, n3 = Hc * SHI;
        cvt_f32_bf16_kernel<<<(n1 / 8 + 255) / 256, 256, 0, stream>>>(x, xbf, n1);
        cvt_f32_bf16_kernel<<<(n2 / 8 + 255) / 256, 256, 0, stream>>>(shup, wupbf, n2);
        cvt_f32_bf16_kernel<<<(n3 / 8 + 255) / 256, 256, 0, stream>>>(shdn, wdnbf, n3);
    }

    gate_kernel<<<Tc, Ec, 0, stream>>>(x, gate_w, gbias, ecnt, etok, ewt);

    shared_up_kernel<<<dim3(SHI / 64, Tc / 16), 128, 0, stream>>>(xbf, wupbf, act);
    shared_down_kernel<<<dim3(Hc / 64, Tc / 16), 128, 0, stream>>>(act, wdnbf, out);

    routed_kernel<<<dim3(Tc / 16, Ec), 256, 0, stream>>>(xbf, w1, w2,
                                                         ecnt, etok, ewt, out);
}
